// AttentionPairBias_75900662055649
// MI455X (gfx1250) — matrix operand order repaired and checked by reading, not compiled, not run on hardware
//
#include <hip/hip_runtime.h>
#include <hip/hip_bf16.h>

// ---------------------------------------------------------------------------
// AttentionPairBias for MI455X (gfx1250, wave32, WMMA f16 16x16x32)
// B=1, N=768, H=16, DH=32, D=512, CZ=128
// ---------------------------------------------------------------------------

typedef __attribute__((ext_vector_type(16))) _Float16 v16h;
typedef __attribute__((ext_vector_type(8)))  _Float16 v8h;
typedef __attribute__((ext_vector_type(8)))  float    v8f;

#define WMMA_F16(A, B, C) \
    __builtin_amdgcn_wmma_f32_16x16x32_f16(false, (A), false, (B), (short)0, (C), false, false)

__device__ __forceinline__ int laneId() { return threadIdx.x & 31; }

// A-fragment (16x32 f16, MxK). tile -> row-major [16][ld] at tile origin.
// Lane l (l<16): row=l,   K = 0..7 (elem 0-7), 16..23 (elem 8-15)
// Lane l (l>=16): row=l-16, K = 8..15, 24..31
__device__ __forceinline__ v16h load_fragA(const _Float16* tile, int ld) {
    const int l   = laneId();
    const int row = l & 15;
    const int ko  = (l >> 4) << 3;
    const _Float16* p = tile + (long)row * ld + ko;
    v8h lo = *(const v8h*)p;
    v8h hi = *(const v8h*)(p + 16);
    v16h r;
#pragma unroll
    for (int i = 0; i < 8; ++i) { r[i] = lo[i]; r[i + 8] = hi[i]; }
    return r;
}

// B-fragment (32x16 f16, KxN) from TRANSPOSED storage tileT = [N][ld] (row n holds B[:,n]).
// Lane l: col n = l&15 ; K = 8*(l>>4) + 0..7 (elem 0-7), 16 + 8*(l>>4) + 0..7 (elem 8-15)
__device__ __forceinline__ v16h load_fragB(const _Float16* tileT, int ld) {
    const int l = laneId();
    const _Float16* p = tileT + (long)(l & 15) * ld + ((l >> 4) << 3);
    v8h lo = *(const v8h*)p;
    v8h hi = *(const v8h*)(p + 16);
    v16h r;
#pragma unroll
    for (int i = 0; i < 8; ++i) { r[i] = lo[i]; r[i + 8] = hi[i]; }
    return r;
}

// ---------------------------------------------------------------------------
// Kernel 1: transpose + cast the 5 [512x512] f32 weight matrices to f16 [N][K]
// ---------------------------------------------------------------------------
__global__ void wconv_kernel(const float* __restrict__ qw, const float* __restrict__ kw,
                             const float* __restrict__ vw, const float* __restrict__ gw,
                             const float* __restrict__ ow,
                             _Float16* qT, _Float16* kT, _Float16* vT, _Float16* gT, _Float16* oT) {
    long idx = (long)blockIdx.x * 256 + threadIdx.x;      // 5 * 512 * 512 total
    int mat = (int)(idx >> 18);
    int k   = (int)(idx >> 9) & 511;
    int n   = (int)idx & 511;
    const float* W = (mat == 0) ? qw : (mat == 1) ? kw : (mat == 2) ? vw : (mat == 3) ? gw : ow;
    _Float16*    T = (mat == 0) ? qT : (mat == 1) ? kT : (mat == 2) ? vT : (mat == 3) ? gT : oT;
    T[(long)n * 512 + k] = (_Float16)W[(long)k * 512 + n];
}

// ---------------------------------------------------------------------------
// Kernel 2: LayerNorm(s) -> f16   (one block per row of 512)
// ---------------------------------------------------------------------------
__global__ void ln_s_kernel(const float* __restrict__ s, const float* __restrict__ w,
                            const float* __restrict__ b, _Float16* __restrict__ out) {
    __shared__ float rs[256], rq[256];
    const int row = blockIdx.x, tid = threadIdx.x;
    float x0 = s[(long)row * 512 + tid];
    float x1 = s[(long)row * 512 + 256 + tid];
    rs[tid] = x0 + x1;
    rq[tid] = x0 * x0 + x1 * x1;
    __syncthreads();
    for (int off = 128; off > 0; off >>= 1) {
        if (tid < off) { rs[tid] += rs[tid + off]; rq[tid] += rq[tid + off]; }
        __syncthreads();
    }
    float mu  = rs[0] * (1.f / 512.f);
    float var = rq[0] * (1.f / 512.f) - mu * mu;
    float inv = rsqrtf(var + 1e-5f);
    out[(long)row * 512 + tid]       = (_Float16)((x0 - mu) * inv * w[tid] + b[tid]);
    out[(long)row * 512 + 256 + tid] = (_Float16)((x1 - mu) * inv * w[tid + 256] + b[tid + 256]);
}

// ---------------------------------------------------------------------------
// Kernel 3: fused Q/K/V/G projections. One wave computes a 16x64 tile:
// the A-fragment is loaded once per k-step and reused by 4 WMMAs.
//   mat 0: q16 = sn@qw + qb     (f16, [768][512])
//   mat 1: k16 = sn@kw          (f16, [768][512])
//   mat 2: vT  = (sn@vw)^T      (f16, [512][768], row = h*32+d)
//   mat 3: g   = sigmoid(sn@gw) (f32, [768][512])
// ---------------------------------------------------------------------------
__global__ void qkvg_gemm_kernel(const _Float16* __restrict__ sn,
                                 const _Float16* __restrict__ qT, const _Float16* __restrict__ kT,
                                 const _Float16* __restrict__ vT, const _Float16* __restrict__ gT,
                                 const float* __restrict__ qb,
                                 _Float16* q16, _Float16* k16, _Float16* vTo, float* g) {
    const int wave = threadIdx.x >> 5;
    const int job  = blockIdx.x * 8 + wave;          // 0..1535, exact
    const int mat  = job / 384;                       // 4 matrices
    const int rem  = job - mat * 384;
    const int mt   = rem >> 3;                        // 48 M-tiles
    const int ng   = rem & 7;                         // 8 groups of 64 columns
    const _Float16* wT = (mat == 0) ? qT : (mat == 1) ? kT : (mat == 2) ? vT : gT;

    v8f acc[4] = {{}, {}, {}, {}};
    const _Float16* abase = sn + (long)mt * 16 * 512;
    const _Float16* bbase = wT + (long)ng * 64 * 512;
#pragma unroll 2
    for (int k0 = 0; k0 < 512; k0 += 32) {
        v16h a = load_fragA(abase + k0, 512);
#pragma unroll
        for (int t = 0; t < 4; ++t) {
            v16h b = load_fragB(bbase + (long)t * 16 * 512 + k0, 512);
            acc[t] = WMMA_F16(a, b, acc[t]);
        }
    }
    const int l  = laneId();
    const int n  = l & 15;
    const int mb = (l >> 4) << 3;
#pragma unroll
    for (int t = 0; t < 4; ++t) {
        const int col = ng * 64 + t * 16 + n;
#pragma unroll
        for (int r = 0; r < 8; ++r) {
            int   row = mt * 16 + mb + r;
            float x   = acc[t][r];
            if (mat == 0)      q16[(long)row * 512 + col] = (_Float16)(x + qb[col]);
            else if (mat == 1) k16[(long)row * 512 + col] = (_Float16)x;
            else if (mat == 2) vTo[(long)col * 768 + row] = (_Float16)x;
            else               g[(long)row * 512 + col]   = __builtin_amdgcn_rcpf(1.f + __expf(-x));
        }
    }
}

// ---------------------------------------------------------------------------
// Kernel 4: pair bias: zb[h][i][j] = (layernorm_CZ(z[i][j]) @ z_w)[h]
// One wave per (i,j): single coalesced 512B read of z, shfl mean/var,
// LDS staging for the 128->16 dot. Reads z exactly once (302 MB = the roofline).
// ---------------------------------------------------------------------------
__global__ void zbias_kernel(const float* __restrict__ z, const float* __restrict__ znw,
                             const float* __restrict__ znb, const float* __restrict__ zw,
                             float* __restrict__ zb) {
    __shared__ float zwsh[128 * 16];
    __shared__ float nvsh[8][128];
    const int tid = threadIdx.x;
    for (int t = tid; t < 2048; t += 256) zwsh[t] = zw[t];
    __syncthreads();

    const int wave = tid >> 5, l = tid & 31;
    const long pair = (long)blockIdx.x * 8 + wave;    // exact: 768*768
    const int i = (int)(pair / 768), j = (int)(pair % 768);

    const float4 zv = *(const float4*)(z + ((long)i * 768 + j) * 128 + l * 4);
    float sum = zv.x + zv.y + zv.z + zv.w;
    float sq  = zv.x * zv.x + zv.y * zv.y + zv.z * zv.z + zv.w * zv.w;
#pragma unroll
    for (int off = 16; off > 0; off >>= 1) {
        sum += __shfl_xor(sum, off);
        sq  += __shfl_xor(sq, off);
    }
    float mu  = sum * (1.f / 128.f);
    float var = sq * (1.f / 128.f) - mu * mu;
    float inv = rsqrtf(var + 1e-5f);
    const int c = l * 4;
    nvsh[wave][c + 0] = (zv.x - mu) * inv * znw[c + 0] + znb[c + 0];
    nvsh[wave][c + 1] = (zv.y - mu) * inv * znw[c + 1] + znb[c + 1];
    nvsh[wave][c + 2] = (zv.z - mu) * inv * znw[c + 2] + znb[c + 2];
    nvsh[wave][c + 3] = (zv.w - mu) * inv * znw[c + 3] + znb[c + 3];
    asm volatile("s_wait_dscnt 0" ::: "memory");      // wave-local LDS RAW
    if (l < 16) {
        float acc = 0.f;
#pragma unroll 8
        for (int cc = 0; cc < 128; ++cc) acc += nvsh[wave][cc] * zwsh[cc * 16 + l];
        zb[((long)l * 768 + i) * 768 + j] = acc;
    }
}

// ---------------------------------------------------------------------------
// Kernel 5: flash attention. One wave per (head, 16-row q-tile); 4 waves per
// block share the head, so K/V 32x32 tiles are staged ONCE per block into LDS
// with global_load_async_to_lds_b128 (ASYNCcnt), then consumed as WMMA B-frags.
// Online softmax with 16-lane shfl row reductions; P transposed via LDS.
// ---------------------------------------------------------------------------
__global__ void attn_kernel(const _Float16* __restrict__ q16, const _Float16* __restrict__ k16,
                            const _Float16* __restrict__ vT16, const float* __restrict__ zb,
                            float* __restrict__ o) {
    __shared__ _Float16 psh[4][16 * 32];
    __shared__ _Float16 ksh[32 * 32];                 // [j_local][d]   (B^T for QK^T)
    __shared__ _Float16 vsh[32 * 32];                 // [d][j_local]   (B^T for PV)
    const int l = laneId(), wave = threadIdx.x >> 5;
    const int job = blockIdx.x * 4 + wave;            // 0..767 exact; same h per block
    const int h = job / 48, qt = job % 48;
    const float scale = 0.17677669529663687f;         // 32^-0.5

    const v16h aq = load_fragA(q16 + (long)(qt * 16) * 512 + h * 32, 512);

    float m8[8], l8[8];
    v8f o0 = {}, o1 = {};
#pragma unroll
    for (int r = 0; r < 8; ++r) { m8[r] = -1e30f; l8[r] = 0.f; }

    const int ncol  = l & 15;
    const int mb    = (l >> 4) << 3;
    const int mrow0 = qt * 16 + mb;                   // global q row of this lane's vgpr 0
    _Float16* pw = &psh[wave][0];

    // cooperative async staging: wave w stages rows w*8..w*8+7 of each 32x32 tile,
    // lane -> (row = w*8 + l/4, 16B chunk = l%4)
    const int srow = wave * 8 + (l >> 2);
    const int sch  = (l & 3) * 8;                     // chunk offset in halves
    const unsigned kshOff = (unsigned)(size_t)&ksh[srow * 32 + sch];
    const unsigned vshOff = (unsigned)(size_t)&vsh[srow * 32 + sch];
    const _Float16* kgb = k16  + (long)srow * 512 + h * 32 + sch;          // + j0*512
    const _Float16* vgb = vT16 + (long)(h * 32 + srow) * 768 + sch;        // + j0

    for (int jt = 0; jt < 48; ++jt) {
        const int j0 = jt * 32;
        __syncthreads();                              // prior tile fully consumed
        const _Float16* kg = kgb + (long)j0 * 512;
        const _Float16* vg = vgb + j0;
        asm volatile("global_load_async_to_lds_b128 %0, %1, off"
                     :: "v"(kshOff), "v"(kg) : "memory");
        asm volatile("global_load_async_to_lds_b128 %0, %1, off"
                     :: "v"(vshOff), "v"(vg) : "memory");
        asm volatile("s_wait_asynccnt 0" ::: "memory");
        __syncthreads();                              // tiles visible to all waves

        v16h bk0 = load_fragB(&ksh[0], 32);
        v16h bk1 = load_fragB(&ksh[16 * 32], 32);
        v8f z8 = {};
        v8f s0 = WMMA_F16(aq, bk0, z8);
        v8f s1 = WMMA_F16(aq, bk1, z8);
#pragma unroll
        for (int r = 0; r < 8; ++r) {
            const long zrow = (long)h * 768 + (mrow0 + r);
            float b0 = zb[zrow * 768 + j0 + ncol];
            float b1 = zb[zrow * 768 + j0 + 16 + ncol];
            float v0 = s0[r] * scale + b0;
            float v1 = s1[r] * scale + b1;
            // row max over 32 cols: elementwise max then 16-lane xor reduce
            float t = fmaxf(v0, v1);
            t = fmaxf(t, __shfl_xor(t, 1));
            t = fmaxf(t, __shfl_xor(t, 2));
            t = fmaxf(t, __shfl_xor(t, 4));
            t = fmaxf(t, __shfl_xor(t, 8));
            float mn   = fmaxf(m8[r], t);
            float corr = __expf(m8[r] - mn);
            float p0   = __expf(v0 - mn);
            float p1   = __expf(v1 - mn);
            float rs = p0 + p1;
            rs += __shfl_xor(rs, 1);
            rs += __shfl_xor(rs, 2);
            rs += __shfl_xor(rs, 4);
            rs += __shfl_xor(rs, 8);
            l8[r] = l8[r] * corr + rs;
            m8[r] = mn;
            o0[r] *= corr;
            o1[r] *= corr;
            const int prow = mb + r;
            pw[prow * 32 + ncol]      = (_Float16)p0;
            pw[prow * 32 + 16 + ncol] = (_Float16)p1;
        }
        asm volatile("s_wait_dscnt 0" ::: "memory");  // wave-local LDS transpose RAW
        v16h ap  = load_fragA(pw, 32);
        v16h bv0 = load_fragB(&vsh[0], 32);
        v16h bv1 = load_fragB(&vsh[16 * 32], 32);
        o0 = WMMA_F16(ap, bv0, o0);
        o1 = WMMA_F16(ap, bv1, o1);
    }
#pragma unroll
    for (int r = 0; r < 8; ++r) {
        float inv = 1.f / l8[r];
        long  row = mrow0 + r;
        o[row * 512 + h * 32 + ncol]      = o0[r] * inv;
        o[row * 512 + h * 32 + 16 + ncol] = o1[r] * inv;
    }
}

// ---------------------------------------------------------------------------
// Kernel 6: out = (o * g) @ o_w. 16x64 tile per wave: A built once per k-step
// (cast to f16 on the fly) and reused by 4 WMMAs.
// ---------------------------------------------------------------------------
__global__ void out_gemm_kernel(const float* __restrict__ o, const float* __restrict__ g,
                                const _Float16* __restrict__ owT, float* __restrict__ out) {
    const int wave = threadIdx.x >> 5;
    const int job  = blockIdx.x * 8 + wave;           // 0..383 exact
    const int mt   = job >> 3, ng = job & 7;
    const int l    = laneId();
    const long rb  = (long)(mt * 16 + (l & 15)) * 512;
    const int  ko  = (l >> 4) << 3;

    v8f acc[4] = {{}, {}, {}, {}};
    const _Float16* bbase = owT + (long)ng * 64 * 512;
#pragma unroll 2
    for (int k0 = 0; k0 < 512; k0 += 32) {
        v16h a;
#pragma unroll
        for (int i = 0; i < 8; ++i) {
            long i0 = rb + k0 + ko + i;
            long i1 = i0 + 16;
            a[i]     = (_Float16)(o[i0] * g[i0]);
            a[i + 8] = (_Float16)(o[i1] * g[i1]);
        }
#pragma unroll
        for (int t = 0; t < 4; ++t) {
            v16h b = load_fragB(bbase + (long)t * 16 * 512 + k0, 512);
            acc[t] = WMMA_F16(a, b, acc[t]);
        }
    }
    const int n = l & 15, mb = (l >> 4) << 3;
#pragma unroll
    for (int t = 0; t < 4; ++t)
#pragma unroll
        for (int r = 0; r < 8; ++r)
            out[(long)(mt * 16 + mb + r) * 512 + ng * 64 + t * 16 + n] = acc[t][r];
}

// ---------------------------------------------------------------------------
// Launch
// ---------------------------------------------------------------------------
extern "C" void kernel_launch(void* const* d_in, const int* in_sizes, int n_in,
                              void* d_out, int out_size, void* d_ws, size_t ws_size,
                              hipStream_t stream) {
    const float* s      = (const float*)d_in[0];
    const float* z      = (const float*)d_in[1];
    const float* nsw    = (const float*)d_in[2];
    const float* nsb    = (const float*)d_in[3];
    const float* q_w    = (const float*)d_in[4];
    const float* q_b    = (const float*)d_in[5];
    const float* k_w    = (const float*)d_in[6];
    const float* v_w    = (const float*)d_in[7];
    const float* g_w    = (const float*)d_in[8];
    const float* zn_w   = (const float*)d_in[9];
    const float* zn_b   = (const float*)d_in[10];
    const float* z_w    = (const float*)d_in[11];
    const float* o_w    = (const float*)d_in[12];
    float* out = (float*)d_out;

    char* ws = (char*)d_ws;
    _Float16* sn16 = (_Float16*)(ws);                       // 768K
    _Float16* q16  = (_Float16*)(ws + 786432);
    _Float16* k16  = (_Float16*)(ws + 1572864);
    _Float16* vT16 = (_Float16*)(ws + 2359296);
    float*    g32  = (float*)   (ws + 3145728);             // 1.5M
    _Float16* qT   = (_Float16*)(ws + 4718592);
    _Float16* kT   = (_Float16*)(ws + 5242880);
    _Float16* vT_w = (_Float16*)(ws + 5767168);
    _Float16* gT   = (_Float16*)(ws + 6291456);
    _Float16* oT   = (_Float16*)(ws + 6815744);
    float*    zb   = (float*)   (ws + 7340032);             // 37.75M
    float*    o32  = (float*)   (ws + 45088768);            // 1.5M  (total ~46.7MB)
    (void)in_sizes; (void)n_in; (void)out_size; (void)ws_size;

    wconv_kernel<<<5120, 256, 0, stream>>>(q_w, k_w, v_w, g_w, o_w, qT, kT, vT_w, gT, oT);
    ln_s_kernel<<<768, 256, 0, stream>>>(s, nsw, nsb, sn16);
    qkvg_gemm_kernel<<<192, 256, 0, stream>>>(sn16, qT, kT, vT_w, gT, q_b, q16, k16, vT16, g32);
    zbias_kernel<<<73728, 256, 0, stream>>>(z, zn_w, zn_b, z_w, zb);
    attn_kernel<<<192, 128, 0, stream>>>(q16, k16, vT16, zb, o32);
    out_gemm_kernel<<<48, 256, 0, stream>>>(o32, g32, oT, out);
}
